// Generator_64579128262979
// MI455X (gfx1250) — matrix operand order repaired and checked by reading, not compiled, not run on hardware
//
#include <hip/hip_runtime.h>
#include <hip/hip_bf16.h>
#include <math.h>

// ---------------------------------------------------------------------------
// Problem constants (from the reference)
// ---------------------------------------------------------------------------
#define BB    64
#define SS    512
#define DD    1024
#define HH    512
#define W_SEG 96
#define T_SEG 24
#define T_PAD 32           // tab rows padded to a full WMMA M tile pair
#define C_SEG 128
#define E_TOT 153          // T_SEG + C_SEG + 1
#define E_PAD 160          // padded to 10 WMMA tiles
#define NBINS (W_SEG + T_SEG + C_SEG)   // 248 real bins in LDS

typedef __attribute__((ext_vector_type(16))) _Float16 v16h;
typedef __attribute__((ext_vector_type(8)))  float    v8f;

// ---------------------------------------------------------------------------
// Workspace layout (byte offsets) — seg sums stored directly as f16
// ---------------------------------------------------------------------------
#define OFF_SEGW   ((size_t)0)            // f16 [64, 96,1024] 12,582,912 B
#define OFF_SEGT   ((size_t)12582912)     // f16 [64, 32,1024]  4,194,304 B (rows 24..31 zero)
#define OFF_SEGC   ((size_t)16777216)     // f16 [64,128,1024] 16,777,216 B
#define OFF_WSUBT  ((size_t)33554432)     // f16 [512,1024]     1,048,576 B
#define OFF_WTABT  ((size_t)34603008)
#define OFF_WCOLT  ((size_t)35651584)
#define OFF_SUBENC ((size_t)36700160)     // f16 [64, 96,512]   6,291,456 B
#define OFF_ENTITY ((size_t)42991616)     // f16 [64,160,512]  10,485,760 B
#define OFF_LOGITS ((size_t)53477376)     // f32 [64, 96,160]   3,932,160 B
// total: 57,409,536 bytes

// ---------------------------------------------------------------------------
// 1) Weight convert + transpose: W [D,H] f32 -> Wt [H,D] f16
// ---------------------------------------------------------------------------
__global__ void cvt_transpose_kernel(const float* __restrict__ W, _Float16* __restrict__ Wt) {
  int idx = blockIdx.x * blockDim.x + threadIdx.x;   // 0 .. D*H-1
  if (idx >= DD * HH) return;
  int d = idx & (DD - 1);
  int h = idx >> 10;
  Wt[(size_t)h * DD + d] = (_Float16)W[(size_t)d * HH + h];
}

// ---------------------------------------------------------------------------
// 2) Segment sums: block owns (batch b, 64-wide d tile) exclusively.
//    f32 accumulation in LDS (ds_add_f32), single f16 convert at store.
//    Also writes the zero pad rows of segT (24..31).
// ---------------------------------------------------------------------------
__global__ __launch_bounds__(256)
void segsum_kernel(const float* __restrict__ q,
                   const int* __restrict__ wseg, const int* __restrict__ tseg,
                   const int* __restrict__ cseg,
                   _Float16* __restrict__ segW, _Float16* __restrict__ segT,
                   _Float16* __restrict__ segC) {
  __shared__ float acc[NBINS * 64];                  // 63488 bytes
  const int b     = blockIdx.y;
  const int dbase = blockIdx.x * 64;
  const int tid   = threadIdx.x;

  for (int i = tid; i < NBINS * 64; i += 256) acc[i] = 0.0f;
  __syncthreads();

  const int lane_d = tid & 63;
  const int grp    = tid >> 6;                       // 4 tokens in parallel
  for (int s = grp; s < SS; s += 4) {
    const int ti   = b * SS + s;
    const int wbin = wseg[ti];
    const int tbin = W_SEG + tseg[ti];
    const int cbin = W_SEG + T_SEG + cseg[ti];
    const float v  = q[(size_t)ti * DD + dbase + lane_d];
    atomicAdd(&acc[wbin * 64 + lane_d], v);          // ds_add_f32
    atomicAdd(&acc[tbin * 64 + lane_d], v);
    atomicAdd(&acc[cbin * 64 + lane_d], v);
  }
  __syncthreads();

  // 256 logical output bins: 96 word + 32 tab(padded) + 128 col
  for (int i = tid; i < 256 * 64; i += 256) {
    const int bin = i >> 6;
    const int d   = i & 63;
    if (bin < W_SEG) {
      segW[((size_t)b * W_SEG + bin) * DD + dbase + d] = (_Float16)acc[bin * 64 + d];
    } else if (bin < W_SEG + T_PAD) {
      const int t   = bin - W_SEG;
      const float v = (t < T_SEG) ? acc[(W_SEG + t) * 64 + d] : 0.0f;
      segT[((size_t)b * T_PAD + t) * DD + dbase + d] = (_Float16)v;
    } else {
      const int cc = bin - (W_SEG + T_PAD);
      segC[((size_t)b * C_SEG + cc) * DD + dbase + d] =
          (_Float16)acc[(W_SEG + T_SEG + cc) * 64 + d];
    }
  }
}

// ---------------------------------------------------------------------------
// 3) Encoder GEMM + tanh. One wave computes a 16x64 output strip
//    (4 N-tiles sharing one A fragment) -> 4 WMMAs per K step.
//    A (f16, row-major [Mpad, D]) lane layout:
//      lanes 0-15:  M=lane,    K = k0..k0+15 (contiguous)
//      lanes 16-31: M=lane-16, K = k0+16..k0+31
//    B from Wt [H, D]: lanes 0-15: N=col, K=k0..k0+15 (contiguous);
//                      lanes 16-31: K=k0+16..k0+31.
// ---------------------------------------------------------------------------
__global__ __launch_bounds__(32)
void enc_gemm_kernel(const _Float16* __restrict__ A, const _Float16* __restrict__ Wt,
                     _Float16* __restrict__ out, int M, int rowoff,
                     int aBatch, int oBatch) {
  const int h0    = blockIdx.x * 64;                 // 4 N-tiles per wave
  const int mbase = blockIdx.y * 16;
  const int b     = blockIdx.z;
  const int lane  = threadIdx.x;
  const int half  = lane >> 4;
  const int row   = mbase + (lane & 15);

  const _Float16* Ab = A   + (size_t)b * aBatch;
  _Float16*       Ob = out + (size_t)b * oBatch;

  v8f c0 = {}, c1 = {}, c2 = {}, c3 = {};
  for (int k0 = 0; k0 < DD; k0 += 32) {
    v16h a;
    const _Float16* pa = Ab + (size_t)row * DD + k0 + half * 16;
#pragma unroll
    for (int j = 0; j < 16; ++j) a[j] = pa[j];

    const _Float16* pb = Wt + (size_t)(h0 + (lane & 15)) * DD + k0 + half * 16;
    v16h b0, b1, b2, b3;
#pragma unroll
    for (int j = 0; j < 16; ++j) b0[j] = pb[j];
    pb += (size_t)16 * DD;
#pragma unroll
    for (int j = 0; j < 16; ++j) b1[j] = pb[j];
    pb += (size_t)16 * DD;
#pragma unroll
    for (int j = 0; j < 16; ++j) b2[j] = pb[j];
    pb += (size_t)16 * DD;
#pragma unroll
    for (int j = 0; j < 16; ++j) b3[j] = pb[j];

    c0 = __builtin_amdgcn_wmma_f32_16x16x32_f16(false, a, false, b0, (short)0, c0, false, false);
    c1 = __builtin_amdgcn_wmma_f32_16x16x32_f16(false, a, false, b1, (short)0, c1, false, false);
    c2 = __builtin_amdgcn_wmma_f32_16x16x32_f16(false, a, false, b2, (short)0, c2, false, false);
    c3 = __builtin_amdgcn_wmma_f32_16x16x32_f16(false, a, false, b3, (short)0, c3, false, false);
  }

  // C/D layout: VGPR r, lane l -> M = r + 8*(l>=16), N = l&15
  const int n = lane & 15;
#pragma unroll
  for (int r = 0; r < 8; ++r) {
    const int m = mbase + r + 8 * half;
    if (m < M) {
      _Float16* po = Ob + (size_t)(rowoff + m) * HH + h0 + n;
      po[0]  = (_Float16)tanhf(c0[r]);
      po[16] = (_Float16)tanhf(c1[r]);
      po[32] = (_Float16)tanhf(c2[r]);
      po[48] = (_Float16)tanhf(c3[r]);
    }
  }
}

// ---------------------------------------------------------------------------
// 4) Fill entity row 152 with f16(no_entry) and zero-pad rows 153..159
// ---------------------------------------------------------------------------
__global__ __launch_bounds__(512)
void entity_fix_kernel(const float* __restrict__ ne, _Float16* __restrict__ entity) {
  const int b = blockIdx.x;
  const int h = threadIdx.x;
  _Float16* base = entity + (size_t)b * E_PAD * HH;
  base[(size_t)(T_SEG + C_SEG) * HH + h] = (_Float16)ne[h];
#pragma unroll
  for (int r = E_TOT; r < E_PAD; ++r)
    base[(size_t)r * HH + h] = (_Float16)0.0f;
}

// ---------------------------------------------------------------------------
// 5) Attention GEMM: logits[m,e] = sum_k sub[m,k] * entity[e,k].
//    One wave computes a 16x80 strip (5 N-tiles, shared A fragment).
//    B(k,n) = entity[n,k]: each lane reads 16 contiguous halfs of a row.
// ---------------------------------------------------------------------------
__global__ __launch_bounds__(32)
void attn_gemm_kernel(const _Float16* __restrict__ S, const _Float16* __restrict__ E,
                      float* __restrict__ logits) {
  const int n0    = blockIdx.x * 80;                 // 5 N-tiles per wave
  const int mbase = blockIdx.y * 16;
  const int b     = blockIdx.z;
  const int lane  = threadIdx.x;
  const int half  = lane >> 4;
  const int row   = mbase + (lane & 15);

  const _Float16* Sb = S + (size_t)b * W_SEG * HH;
  const _Float16* Eb = E + (size_t)b * E_PAD * HH;

  v8f c0 = {}, c1 = {}, c2 = {}, c3 = {}, c4 = {};
  for (int k0 = 0; k0 < HH; k0 += 32) {
    v16h a;
    const _Float16* pa = Sb + (size_t)row * HH + k0 + half * 16;
#pragma unroll
    for (int j = 0; j < 16; ++j) a[j] = pa[j];

    const _Float16* pb = Eb + (size_t)(n0 + (lane & 15)) * HH + k0 + half * 16;
    v16h b0, b1, b2, b3, b4;
#pragma unroll
    for (int j = 0; j < 16; ++j) b0[j] = pb[j];
    pb += (size_t)16 * HH;
#pragma unroll
    for (int j = 0; j < 16; ++j) b1[j] = pb[j];
    pb += (size_t)16 * HH;
#pragma unroll
    for (int j = 0; j < 16; ++j) b2[j] = pb[j];
    pb += (size_t)16 * HH;
#pragma unroll
    for (int j = 0; j < 16; ++j) b3[j] = pb[j];
    pb += (size_t)16 * HH;
#pragma unroll
    for (int j = 0; j < 16; ++j) b4[j] = pb[j];

    c0 = __builtin_amdgcn_wmma_f32_16x16x32_f16(false, a, false, b0, (short)0, c0, false, false);
    c1 = __builtin_amdgcn_wmma_f32_16x16x32_f16(false, a, false, b1, (short)0, c1, false, false);
    c2 = __builtin_amdgcn_wmma_f32_16x16x32_f16(false, a, false, b2, (short)0, c2, false, false);
    c3 = __builtin_amdgcn_wmma_f32_16x16x32_f16(false, a, false, b3, (short)0, c3, false, false);
    c4 = __builtin_amdgcn_wmma_f32_16x16x32_f16(false, a, false, b4, (short)0, c4, false, false);
  }

  float* lb = logits + ((size_t)b * W_SEG + mbase) * E_PAD;
  const int n = lane & 15;
#pragma unroll
  for (int r = 0; r < 8; ++r) {
    const int m = r + 8 * half;
    float* po = lb + (size_t)m * E_PAD + n0 + n;
    po[0]  = c0[r];
    po[16] = c1[r];
    po[32] = c2[r];
    po[48] = c3[r];
    po[64] = c4[r];
  }
}

// ---------------------------------------------------------------------------
// 6) Row softmax over the 153 valid entities -> d_out [64,96,153] f32
// ---------------------------------------------------------------------------
__global__ __launch_bounds__(256)
void softmax_kernel(const float* __restrict__ logits, float* __restrict__ out) {
  const int m = blockIdx.x;
  const int b = blockIdx.y;
  const int t = threadIdx.x;
  __shared__ float red[256];

  const float* rowp = logits + ((size_t)b * W_SEG + m) * E_PAD;
  const float v = (t < E_TOT) ? rowp[t] : -INFINITY;

  red[t] = v;
  __syncthreads();
  for (int s = 128; s > 0; s >>= 1) {
    if (t < s) red[t] = fmaxf(red[t], red[t + s]);
    __syncthreads();
  }
  const float mx = red[0];
  __syncthreads();

  const float e = (t < E_TOT) ? __expf(v - mx) : 0.0f;
  red[t] = e;
  __syncthreads();
  for (int s = 128; s > 0; s >>= 1) {
    if (t < s) red[t] += red[t + s];
    __syncthreads();
  }
  const float inv = 1.0f / red[0];

  if (t < E_TOT)
    out[((size_t)b * W_SEG + m) * E_TOT + t] = e * inv;
}

// ---------------------------------------------------------------------------
// Launch
// ---------------------------------------------------------------------------
extern "C" void kernel_launch(void* const* d_in, const int* in_sizes, int n_in,
                              void* d_out, int out_size, void* d_ws, size_t ws_size,
                              hipStream_t stream) {
  const float* q_enc = (const float*)d_in[0];
  const float* W_sub = (const float*)d_in[1];
  const float* W_tab = (const float*)d_in[2];
  const float* W_col = (const float*)d_in[3];
  const float* ne    = (const float*)d_in[4];
  const int*   wseg  = (const int*)d_in[5];
  const int*   tseg  = (const int*)d_in[6];
  const int*   cseg  = (const int*)d_in[7];

  char* ws = (char*)d_ws;
  _Float16*  segW   = (_Float16*)(ws + OFF_SEGW);
  _Float16*  segT   = (_Float16*)(ws + OFF_SEGT);
  _Float16*  segC   = (_Float16*)(ws + OFF_SEGC);
  _Float16*  WsubT  = (_Float16*)(ws + OFF_WSUBT);
  _Float16*  WtabT  = (_Float16*)(ws + OFF_WTABT);
  _Float16*  WcolT  = (_Float16*)(ws + OFF_WCOLT);
  _Float16*  subEnc = (_Float16*)(ws + OFF_SUBENC);
  _Float16*  entity = (_Float16*)(ws + OFF_ENTITY);
  float*     logits = (float*)(ws + OFF_LOGITS);

  // 1) weight f32 -> f16 transpose (3x)
  {
    dim3 g((DD * HH + 255) / 256);
    cvt_transpose_kernel<<<g, 256, 0, stream>>>(W_sub, WsubT);
    cvt_transpose_kernel<<<g, 256, 0, stream>>>(W_tab, WtabT);
    cvt_transpose_kernel<<<g, 256, 0, stream>>>(W_col, WcolT);
  }

  // 2) segment sums (fully overwrites seg buffers incl. tab pad rows)
  segsum_kernel<<<dim3(DD / 64, BB), 256, 0, stream>>>(q_enc, wseg, tseg, cseg,
                                                       segW, segT, segC);

  // 3) encoder GEMMs (tanh + f16 store). tab/col write straight into entity.
  enc_gemm_kernel<<<dim3(HH / 64, W_SEG / 16, BB), 32, 0, stream>>>(
      segW, WsubT, subEnc, W_SEG, 0, W_SEG * DD, W_SEG * HH);
  enc_gemm_kernel<<<dim3(HH / 64, T_PAD / 16, BB), 32, 0, stream>>>(
      segT, WtabT, entity, T_SEG, 0, T_PAD * DD, E_PAD * HH);
  enc_gemm_kernel<<<dim3(HH / 64, C_SEG / 16, BB), 32, 0, stream>>>(
      segC, WcolT, entity, C_SEG, T_SEG, C_SEG * DD, E_PAD * HH);

  // 4) no_entry row + zero padding
  entity_fix_kernel<<<BB, HH, 0, stream>>>(ne, entity);

  // 5) attention logits
  attn_gemm_kernel<<<dim3(E_PAD / 80, W_SEG / 16, BB), 32, 0, stream>>>(
      subEnc, entity, logits);

  // 6) softmax -> output [64, 96, 153] f32
  softmax_kernel<<<dim3(W_SEG, BB), 256, 0, stream>>>(logits, (float*)d_out);
}
